// SSP_SAM_26688926777355
// MI455X (gfx1250) — matrix operand order repaired and checked by reading, not compiled, not run on hardware
//
#include <hip/hip_runtime.h>
#include <hip/hip_bf16.h>

typedef __bf16 bf16;
typedef bf16  v16bf __attribute__((ext_vector_type(16)));
typedef bf16  v8bf  __attribute__((ext_vector_type(8)));
typedef bf16  v4bf  __attribute__((ext_vector_type(4)));
typedef float v8f   __attribute__((ext_vector_type(8)));

#define WMMA_BF16(a, b, c) \
  __builtin_amdgcn_wmma_f32_16x16x32_bf16(false, (a), false, (b), (short)0, (c), false, false)

// Async global->LDS copy of 16 bytes per lane (CDNA5, tracked by ASYNCcnt).
// vdst = per-lane LDS byte address (low 32 bits of generic pointer), vaddr = global addr.
__device__ __forceinline__ void async_ld_b128(unsigned lds_addr, const void* gaddr) {
  asm volatile("global_load_async_to_lds_b128 %0, %1, off"
               :: "v"(lds_addr), "v"(gaddr)
               : "memory");
}
__device__ __forceinline__ void async_wait0() {
  asm volatile("s_wait_asynccnt 0x0" ::: "memory");
}
__device__ __forceinline__ unsigned lds_off(const void* p) {
  return (unsigned)(size_t)p;  // flat LDS aperture: low 32 bits == LDS byte offset
}

// ---------------------------------------------------------------------------
// Fragment loaders from LDS (bf16, 16x16x32 WMMA).
// A: 16x32 (MxK). lane lo = row M; elems 0..7 -> K = hi*8+0..7, 8..15 -> K = 16+hi*8+0..7
// B: 32x16 (KxN) loaded as rows of B^T. lane lo = N; elems 0..7 -> K = hi*8+0..7, 8..15 -> K = 16+hi*8+0..7
// ---------------------------------------------------------------------------
__device__ __forceinline__ v16bf frag_a(const bf16* base, int stride) {
  int lane = threadIdx.x & 31;
  int lo = lane & 15, hi = lane >> 4;
  const bf16* p = base + lo * stride + hi * 8;
  v8bf c0 = *(const v8bf*)(p);
  v8bf c1 = *(const v8bf*)(p + 16);
  v16bf r;
#pragma unroll
  for (int i = 0; i < 8; ++i) { r[i] = c0[i]; r[i + 8] = c1[i]; }
  return r;
}

__device__ __forceinline__ v16bf frag_b(const bf16* base, int stride) {
  int lane = threadIdx.x & 31;
  int lo = lane & 15, hi = lane >> 4;
  const bf16* p = base + lo * stride + hi * 8;
  v8bf c0 = *(const v8bf*)(p);
  v8bf c1 = *(const v8bf*)(p + 16);
  v16bf r;
#pragma unroll
  for (int i = 0; i < 8; ++i) { r[i] = c0[i]; r[i + 8] = c1[i]; }
  return r;
}

// ---------------------------------------------------------------------------
// f32 -> bf16 conversion
// ---------------------------------------------------------------------------
__global__ __launch_bounds__(256) void cvt_f32_bf16(const float* __restrict__ src,
                                                    bf16* __restrict__ dst, long n) {
  long i = ((long)blockIdx.x * blockDim.x + threadIdx.x) * 4;
  if (i >= n) return;
  float4 v = *(const float4*)(src + i);
  v4bf o = { (bf16)v.x, (bf16)v.y, (bf16)v.z, (bf16)v.w };
  *(v4bf*)(dst + i) = o;
}

// ---------------------------------------------------------------------------
// k_posy / k_posx : (29,1024) = pos @ wk-slices. Tiny; f32 VALU, bf16 out.
// Layout: [h][p(pad32)][u] bf16
// ---------------------------------------------------------------------------
__global__ __launch_bounds__(256) void kpos_kernel(const float* __restrict__ pos_y,
                                                   const float* __restrict__ pos_x,
                                                   const float* __restrict__ ipw,
                                                   bf16* __restrict__ kpyB,
                                                   bf16* __restrict__ kpxB) {
  int i = blockIdx.x * blockDim.x + threadIdx.x;
  if (i >= 29 * 1024) return;
  int p = i >> 10, col = i & 1023;
  const float* wk = ipw + (size_t)(1024 + col) * 1024;  // wk row `col`
  float ay = 0.f, ax = 0.f;
  for (int q = 0; q < 512; ++q) {
    ay += pos_y[p * 512 + q] * wk[q];
    ax += pos_x[p * 512 + q] * wk[512 + q];
  }
  int h = col >> 6, u = col & 63;
  size_t d = ((size_t)h * 32 + p) * 64 + u;
  kpyB[d] = (bf16)ay;
  kpxB[d] = (bf16)ax;
}

// ---------------------------------------------------------------------------
// Mask cumsums: yy = cumsum over rows, xx = cumsum over cols of !mask
// ---------------------------------------------------------------------------
__global__ __launch_bounds__(224) void cumsum_kernel(const unsigned char* __restrict__ mask,
                                                     float* __restrict__ yy,
                                                     float* __restrict__ xx) {
  int b = blockIdx.x, g = threadIdx.x;
  if (g >= 196) return;
  int r = g / 14, c = g % 14;
  float sy = 0.f, sx = 0.f;
  for (int rr = 0; rr <= r; ++rr) sy += mask[b * 196 + rr * 14 + c] ? 0.f : 1.f;
  for (int cc = 0; cc <= c; ++cc) sx += mask[b * 196 + r * 14 + cc] ? 0.f : 1.f;
  yy[b * 196 + g] = sy;
  xx[b * 196 + g] = sx;
}

// ---------------------------------------------------------------------------
// WMMA GEMM: Y = X @ W^T + bias. M=12544, N=1024, K=1024.
// Block 128x128, 256 threads = 8 waves; wave = 32(M) x 64(N) = 2x4 tiles.
// Double-buffered LDS ping-pong, 1 barrier per K-step, global loads 2 stages ahead.
// mode 0: q  -> qb [b,h,t(208),dk]  (scaled 0.125)
// mode 1: k  -> kb [b,h,t(208),dk]
// mode 2: v  -> vT [b,h,dk,t(224)]
// mode 3: out-proj -> f32 d_out [t*64+b][h*64+u]
// ---------------------------------------------------------------------------
#define LDA 48

__global__ __launch_bounds__(256) void gemm_bf16(const bf16* __restrict__ Xp,
                                                 const bf16* __restrict__ Wp,
                                                 const float* __restrict__ biasp,
                                                 int mode_base,
                                                 bf16* __restrict__ qb,
                                                 bf16* __restrict__ kb,
                                                 bf16* __restrict__ vT,
                                                 float* __restrict__ outF) {
  const int mb = blockIdx.x;                 // 0..97
  const int nb = blockIdx.y;                 // 0..7
  const int mode = mode_base + blockIdx.z;
  const bf16* X = Xp + (size_t)blockIdx.z * (12544u * 1024u);
  const bf16* W = Wp + (size_t)blockIdx.z * (1024u * 1024u);
  const float* bias = biasp + blockIdx.z * 1024;

  __shared__ bf16 At[2][128 * LDA];
  __shared__ bf16 Bt[2][128 * LDA];

  const int tid = threadIdx.x;
  const int wave = tid >> 5;
  const int wm = wave & 3, wn = wave >> 2;

  // Per-thread staging chunks: 512 chunks of 8 bf16 per 128x32 tile, 2 per thread.
  const int c0 = tid, c1 = tid + 256;
  const int r0 = c0 >> 2, kk0 = (c0 & 3) * 8;
  const int r1 = c1 >> 2, kk1 = (c1 & 3) * 8;
  const bf16* Xr0 = &X[(size_t)(mb * 128 + r0) * 1024 + kk0];
  const bf16* Xr1 = &X[(size_t)(mb * 128 + r1) * 1024 + kk1];
  const bf16* Wr0 = &W[(size_t)(nb * 128 + r0) * 1024 + kk0];
  const bf16* Wr1 = &W[(size_t)(nb * 128 + r1) * 1024 + kk1];

  // stage 0 -> LDS buf 0
  {
    v8bf a0 = *(const v8bf*)(Xr0), a1 = *(const v8bf*)(Xr1);
    v8bf b0 = *(const v8bf*)(Wr0), b1 = *(const v8bf*)(Wr1);
    *(v8bf*)&At[0][r0 * LDA + kk0] = a0;
    *(v8bf*)&At[0][r1 * LDA + kk1] = a1;
    *(v8bf*)&Bt[0][r0 * LDA + kk0] = b0;
    *(v8bf*)&Bt[0][r1 * LDA + kk1] = b1;
  }
  // stage 1 prefetch -> VGPRs
  v8bf pa0 = *(const v8bf*)(Xr0 + 32), pa1 = *(const v8bf*)(Xr1 + 32);
  v8bf pb0 = *(const v8bf*)(Wr0 + 32), pb1 = *(const v8bf*)(Wr1 + 32);
  __syncthreads();

  v8f acc[2][4] = {};

  for (int k0 = 0; k0 < 1024; k0 += 32) {
    const int cur = (k0 >> 5) & 1;
    if (k0 + 32 < 1024) {
      // store next stage into other buffer
      *(v8bf*)&At[cur ^ 1][r0 * LDA + kk0] = pa0;
      *(v8bf*)&At[cur ^ 1][r1 * LDA + kk1] = pa1;
      *(v8bf*)&Bt[cur ^ 1][r0 * LDA + kk0] = pb0;
      *(v8bf*)&Bt[cur ^ 1][r1 * LDA + kk1] = pb1;
      if (k0 + 64 < 1024) {
        pa0 = *(const v8bf*)(Xr0 + k0 + 64);
        pa1 = *(const v8bf*)(Xr1 + k0 + 64);
        pb0 = *(const v8bf*)(Wr0 + k0 + 64);
        pb1 = *(const v8bf*)(Wr1 + k0 + 64);
      }
    }

    v16bf a0 = frag_a(&At[cur][(wm * 32 + 0) * LDA], LDA);
    v16bf a1 = frag_a(&At[cur][(wm * 32 + 16) * LDA], LDA);
#pragma unroll
    for (int nt = 0; nt < 4; ++nt) {
      v16bf b = frag_b(&Bt[cur][(wn * 64 + nt * 16) * LDA], LDA);
      acc[0][nt] = WMMA_BF16(a0, b, acc[0][nt]);
      acc[1][nt] = WMMA_BF16(a1, b, acc[1][nt]);
    }
    __syncthreads();
  }

  const int lane = tid & 31, lo = lane & 15, hi = lane >> 4;
#pragma unroll
  for (int mt = 0; mt < 2; ++mt) {
#pragma unroll
    for (int nt = 0; nt < 4; ++nt) {
#pragma unroll
      for (int j = 0; j < 8; ++j) {
        int gr = mb * 128 + wm * 32 + mt * 16 + j + hi * 8;
        int gc = nb * 128 + wn * 64 + nt * 16 + lo;
        float v = acc[mt][nt][j] + bias[gc];
        int t = gr >> 6, bch = gr & 63;
        int h = gc >> 6, u = gc & 63;
        if (mode == 0) {
          qb[(((size_t)(bch * 16 + h)) * 208 + t) * 64 + u] = (bf16)(v * 0.125f);
        } else if (mode == 1) {
          kb[(((size_t)(bch * 16 + h)) * 208 + t) * 64 + u] = (bf16)v;
        } else if (mode == 2) {
          vT[(((size_t)(bch * 16 + h)) * 64 + u) * 224 + t] = (bf16)v;
        } else {
          outF[(size_t)gr * 1024 + gc] = v;
        }
      }
    }
  }
}

// ---------------------------------------------------------------------------
// Fused attention: one workgroup per (b,h). 256 threads = 8 waves.
// S = Q K^T + gather(posy_w, posx_w) + mask ; softmax ; O = P V
// Prologue LDS fill uses CDNA5 async global->LDS (ASYNCcnt).
// ---------------------------------------------------------------------------
__global__ __launch_bounds__(256) void attn_kernel(const bf16* __restrict__ qb,
                                                   const bf16* __restrict__ kb,
                                                   const bf16* __restrict__ vT,
                                                   const bf16* __restrict__ kpyB,
                                                   const bf16* __restrict__ kpxB,
                                                   const float* __restrict__ yyB,
                                                   const float* __restrict__ xxB,
                                                   const unsigned char* __restrict__ kmask,
                                                   bf16* __restrict__ X2) {
  const int head = blockIdx.x;               // b*16 + h
  const int bch = head >> 4, h = head & 15;

  __shared__ bf16 Qs[208 * 72];
  __shared__ bf16 Ks[208 * 72];
  __shared__ bf16 Vs[64 * 232];
  __shared__ bf16 Py[32 * 64];
  __shared__ bf16 Px[32 * 64];
  __shared__ float yS[224], xS[224], mS[224];
  __shared__ __align__(16) unsigned char wscr[8][7168];  // per-wave scratch

  const int tid = threadIdx.x;
  const int wave = tid >> 5, lane = tid & 31;
  const int lo = lane & 15, hi = lane >> 4;

  // ---- async bulk LDS fill (16B chunks, ASYNCcnt-tracked) ----
  const bf16* qg = qb + (size_t)head * 208 * 64;
  const bf16* kg = kb + (size_t)head * 208 * 64;
  for (int i = tid; i < 208 * 8; i += 256) {
    int r = i >> 3, kk = (i & 7) * 8;
    async_ld_b128(lds_off(&Qs[r * 72 + kk]), &qg[r * 64 + kk]);
    async_ld_b128(lds_off(&Ks[r * 72 + kk]), &kg[r * 64 + kk]);
  }
  const bf16* vg = vT + (size_t)head * 64 * 224;
  for (int i = tid; i < 64 * 28; i += 256) {
    int r = i / 28, kk = (i % 28) * 8;
    async_ld_b128(lds_off(&Vs[r * 232 + kk]), &vg[r * 224 + kk]);
  }
  for (int i = tid; i < 256; i += 256) {
    async_ld_b128(lds_off(&Py[i * 8]), &kpyB[(size_t)h * 2048 + i * 8]);
    async_ld_b128(lds_off(&Px[i * 8]), &kpxB[(size_t)h * 2048 + i * 8]);
  }
  // small scalar-ish fills stay on the normal path
  for (int i = tid; i < 224; i += 256) {
    bool valid = i < 196;
    yS[i] = valid ? yyB[bch * 196 + i] : 0.f;
    xS[i] = valid ? xxB[bch * 196 + i] : 0.f;
    mS[i] = (valid && !kmask[bch * 196 + i]) ? 0.f : -1e30f;
  }
  async_wait0();
  __syncthreads();

  float* posy = (float*)wscr[wave];          // 16x32 f32
  float* posx = posy + 512;                  // 16x32 f32
  bf16* Pst = (bf16*)wscr[wave];             // 16x224 bf16 (reuses pos space later)

  for (int strip = wave; strip < 13; strip += 8) {
    const int m0 = strip * 16;
    v16bf aq0 = frag_a(&Qs[m0 * 72], 72);
    v16bf aq1 = frag_a(&Qs[m0 * 72 + 32], 72);

    // positional weights: Qstrip @ kpos^T  (16x64 @ 64x32)
    v8f py0 = {}, py1 = {}, px0 = {}, px1 = {};
    py0 = WMMA_BF16(aq0, frag_b(&Py[0], 64), py0);
    py0 = WMMA_BF16(aq1, frag_b(&Py[32], 64), py0);
    py1 = WMMA_BF16(aq0, frag_b(&Py[16 * 64], 64), py1);
    py1 = WMMA_BF16(aq1, frag_b(&Py[16 * 64 + 32], 64), py1);
    px0 = WMMA_BF16(aq0, frag_b(&Px[0], 64), px0);
    px0 = WMMA_BF16(aq1, frag_b(&Px[32], 64), px0);
    px1 = WMMA_BF16(aq0, frag_b(&Px[16 * 64], 64), px1);
    px1 = WMMA_BF16(aq1, frag_b(&Px[16 * 64 + 32], 64), px1);
#pragma unroll
    for (int j = 0; j < 8; ++j) {
      int ml = j + hi * 8;
      posy[ml * 32 + lo] = py0[j];
      posy[ml * 32 + 16 + lo] = py1[j];
      posx[ml * 32 + lo] = px0[j];
      posx[ml * 32 + 16 + lo] = px1[j];
    }

    // S = Q K^T
    v8f s[13] = {};
#pragma unroll
    for (int nt = 0; nt < 13; ++nt) {
      s[nt] = WMMA_BF16(aq0, frag_b(&Ks[nt * 16 * 72], 72), s[nt]);
      s[nt] = WMMA_BF16(aq1, frag_b(&Ks[nt * 16 * 72 + 32], 72), s[nt]);
    }

    // relative-position gather + key mask
#pragma unroll
    for (int nt = 0; nt < 13; ++nt) {
      int n = nt * 16 + lo;
      int nn = n < 196 ? n : 195;
      float pen = mS[n];
#pragma unroll
      for (int j = 0; j < 8; ++j) {
        int ml = j + hi * 8;
        int m = m0 + ml;
        int tm = m < 196 ? m : 195;
        int idy = (int)(yS[tm] - yS[nn] + 14.5f);
        int idx = (int)(xS[tm] - xS[nn] + 14.5f);
        s[nt][j] += posy[ml * 32 + idy] + posx[ml * 32 + idx] + pen;
      }
    }

    // row-wise softmax (rows live across lo-lanes of each 16-lane half + 13 accs)
#pragma unroll
    for (int j = 0; j < 8; ++j) {
      float mx = -3.0e38f;
#pragma unroll
      for (int nt = 0; nt < 13; ++nt) mx = fmaxf(mx, s[nt][j]);
      for (int o = 8; o; o >>= 1) mx = fmaxf(mx, __shfl_xor(mx, o, 16));
      float sum = 0.f;
#pragma unroll
      for (int nt = 0; nt < 13; ++nt) {
        float e = __expf(s[nt][j] - mx);
        s[nt][j] = e;
        sum += e;
      }
      for (int o = 8; o; o >>= 1) sum += __shfl_xor(sum, o, 16);
      float inv = 1.0f / sum;
#pragma unroll
      for (int nt = 0; nt < 13; ++nt) s[nt][j] *= inv;
    }

    // P strip to wave-private LDS (bf16), pad cols 208..223 with zero
#pragma unroll
    for (int nt = 0; nt < 13; ++nt)
#pragma unroll
      for (int j = 0; j < 8; ++j)
        Pst[(j + hi * 8) * 224 + nt * 16 + lo] = (bf16)s[nt][j];
    {
      bf16* z = &Pst[lo * 224 + 208 + hi * 8];
#pragma unroll
      for (int i = 0; i < 8; ++i) z[i] = (bf16)0.f;
    }

    // O = P @ V  (16x224 @ 224x64)
    v8f o[4] = {};
#pragma unroll
    for (int ks = 0; ks < 7; ++ks) {
      v16bf a = frag_a(&Pst[ks * 32], 224);
#pragma unroll
      for (int nt = 0; nt < 4; ++nt)
        o[nt] = WMMA_BF16(a, frag_b(&Vs[nt * 16 * 232 + ks * 32], 232), o[nt]);
    }

    // store to X2[(t*64+b)][h*64+u] (bf16, input to out-proj GEMM)
#pragma unroll
    for (int nt = 0; nt < 4; ++nt)
#pragma unroll
      for (int j = 0; j < 8; ++j) {
        int m = m0 + j + hi * 8;
        if (m < 196) {
          int u = nt * 16 + lo;
          X2[((size_t)m * 64 + bch) * 1024 + h * 64 + u] = (bf16)o[nt][j];
        }
      }
  }
}

// ---------------------------------------------------------------------------
// host: kernel_launch
// ---------------------------------------------------------------------------
extern "C" void kernel_launch(void* const* d_in, const int* in_sizes, int n_in,
                              void* d_out, int out_size, void* d_ws, size_t ws_size,
                              hipStream_t stream) {
  const float* query = (const float*)d_in[0];
  const float* key = (const float*)d_in[1];
  const float* value = (const float*)d_in[2];
  const unsigned char* kmask = (const unsigned char*)d_in[3];
  const float* ipw = (const float*)d_in[4];
  const float* ipb = (const float*)d_in[5];
  const float* outw = (const float*)d_in[6];
  const float* outb = (const float*)d_in[7];
  const float* pos_x = (const float*)d_in[8];
  const float* pos_y = (const float*)d_in[9];

  char* ws = (char*)d_ws;
  size_t o = 0;
  auto alloc = [&](size_t bytes) -> void* {
    o = (o + 255) & ~(size_t)255;
    void* p = ws + o;
    o += bytes;
    return p;
  };

  const long NX = 12544L * 1024L;
  bf16* Xall = (bf16*)alloc(3 * NX * 2);
  bf16* Wbf = (bf16*)alloc(4L * 1024 * 1024 * 2);
  bf16* qbv = (bf16*)alloc(1024L * 208 * 64 * 2);
  bf16* kbv = (bf16*)alloc(1024L * 208 * 64 * 2);
  bf16* vTv = (bf16*)alloc(1024L * 64 * 224 * 2);
  bf16* kpyB = (bf16*)alloc(16L * 32 * 64 * 2);
  bf16* kpxB = (bf16*)alloc(16L * 32 * 64 * 2);
  bf16* X2 = (bf16*)alloc(NX * 2);
  float* yy = (float*)alloc(64L * 196 * 4);
  float* xx = (float*)alloc(64L * 196 * 4);

  // zero padded buffers (t >= 196 rows / p >= 29 rows must be 0)
  hipMemsetAsync(qbv, 0, 1024L * 208 * 64 * 2, stream);
  hipMemsetAsync(kbv, 0, 1024L * 208 * 64 * 2, stream);
  hipMemsetAsync(vTv, 0, 1024L * 64 * 224 * 2, stream);
  hipMemsetAsync(kpyB, 0, 16L * 32 * 64 * 2, stream);
  hipMemsetAsync(kpxB, 0, 16L * 32 * 64 * 2, stream);

  // conversions
  int cb = (int)((NX / 4 + 255) / 256);
  cvt_f32_bf16<<<cb, 256, 0, stream>>>(query, Xall, NX);
  cvt_f32_bf16<<<cb, 256, 0, stream>>>(key, Xall + NX, NX);
  cvt_f32_bf16<<<cb, 256, 0, stream>>>(value, Xall + 2 * NX, NX);
  long NW = 3L * 1024 * 1024;
  cvt_f32_bf16<<<(int)((NW / 4 + 255) / 256), 256, 0, stream>>>(ipw, Wbf, NW);
  long NO = 1024L * 1024;
  cvt_f32_bf16<<<(int)((NO / 4 + 255) / 256), 256, 0, stream>>>(outw, Wbf + NW, NO);

  kpos_kernel<<<(29 * 1024 + 255) / 256, 256, 0, stream>>>(pos_y, pos_x, ipw, kpyB, kpxB);
  cumsum_kernel<<<64, 224, 0, stream>>>(kmask, yy, xx);

  // QKV projections (modes 0,1,2)
  gemm_bf16<<<dim3(98, 8, 3), 256, 0, stream>>>(Xall, Wbf, ipb, 0, qbv, kbv, vTv,
                                                (float*)d_out);

  // fused attention per (b,h)
  attn_kernel<<<1024, 256, 0, stream>>>(qbv, kbv, vTv, kpyB, kpxB, yy, xx, kmask, X2);

  // out projection (mode 3) -> f32 d_out
  gemm_bf16<<<dim3(98, 8, 1), 256, 0, stream>>>(X2, Wbf + NW, outb, 3, qbv, kbv, vTv,
                                                (float*)d_out);
}
